// WeightedBackpackLMHeadModel_12713103196253
// MI455X (gfx1250) — matrix operand order repaired and checked by reading, not compiled, not run on hardware
//
#include <hip/hip_runtime.h>

#define BSZ    2
#define NSENSE 16
#define SEQ    1024
#define DIM    768
#define VOC    50257
#define VPAD   50304   // multiple of 64

typedef __attribute__((ext_vector_type(16))) __bf16 v16bf;
typedef __attribute__((ext_vector_type(8)))  __bf16 v8bf;
typedef __attribute__((ext_vector_type(8)))  float  v8f;

// Native f32 -> bf16 (backend selects hardware cvt; pairs pack to v_cvt_pk_bf16_f32)
static __device__ __forceinline__ __bf16 f2bf(float f) { return (__bf16)f; }

static __device__ __forceinline__ v8f wmma_bf16(v16bf a, v16bf b, v8f c) {
  return __builtin_amdgcn_wmma_f32_16x16x32_bf16(false, a, false, b, (short)0, c,
                                                 false, false);
}

// A fragment (16x32 bf16, MxK): lane = M row (mod 16); per-lane 16 bf16 at
// K offsets {base..base+7, base+16..base+23}, base = lane<16?0:8.
// p points at row + ks + base.
static __device__ __forceinline__ v16bf load_frag_a_bf16(const __bf16* p) {
  v8bf lo = *(const v8bf*)p;
  v8bf hi = *(const v8bf*)(p + 16);
  v16bf a;
#pragma unroll
  for (int t = 0; t < 8; ++t) { a[t] = lo[t]; a[t + 8] = hi[t]; }
  return a;
}

// Same A fragment from an f32 source (convert on load).
static __device__ __forceinline__ v16bf load_frag_a_f32(const float* p) {
  float4 f0 = *(const float4*)(p);
  float4 f1 = *(const float4*)(p + 4);
  float4 f2 = *(const float4*)(p + 16);
  float4 f3 = *(const float4*)(p + 20);
  v16bf a;
  a[0]  = f2bf(f0.x); a[1]  = f2bf(f0.y); a[2]  = f2bf(f0.z); a[3]  = f2bf(f0.w);
  a[4]  = f2bf(f1.x); a[5]  = f2bf(f1.y); a[6]  = f2bf(f1.z); a[7]  = f2bf(f1.w);
  a[8]  = f2bf(f2.x); a[9]  = f2bf(f2.y); a[10] = f2bf(f2.z); a[11] = f2bf(f2.w);
  a[12] = f2bf(f3.x); a[13] = f2bf(f3.y); a[14] = f2bf(f3.z); a[15] = f2bf(f3.w);
  return a;
}

// B fragment (32x16 bf16, KxN): lane = N col (mod 16); per-lane 16 bf16 at
// K offsets {klo..klo+7, klo+16..klo+23}, klo = lane<16?0:8. p points at rowN + ks + klo.
static __device__ __forceinline__ v16bf load_frag_b_bf16(const __bf16* p) {
  v8bf lo = *(const v8bf*)p;
  v8bf hi = *(const v8bf*)(p + 16);
  v16bf b;
#pragma unroll
  for (int t = 0; t < 8; ++t) { b[t] = lo[t]; b[t + 8] = hi[t]; }
  return b;
}

// ---------------- pre/post elementwise kernels ----------------

__global__ void k_convert_w(const float* __restrict__ W, __bf16* __restrict__ Wb) {
  const size_t idx = (size_t)blockIdx.x * blockDim.x + threadIdx.x;  // 8-elem chunk
  const size_t row = idx / (DIM / 8);
  const int    col = (int)(idx % (DIM / 8)) * 8;
  if (row >= (size_t)VPAD) return;
  v8bf o;
  if (row < (size_t)VOC) {
    const float4 f0 = *(const float4*)(W + row * DIM + col);
    const float4 f1 = *(const float4*)(W + row * DIM + col + 4);
    o[0] = f2bf(f0.x); o[1] = f2bf(f0.y); o[2] = f2bf(f0.z); o[3] = f2bf(f0.w);
    o[4] = f2bf(f1.x); o[5] = f2bf(f1.y); o[6] = f2bf(f1.z); o[7] = f2bf(f1.w);
  } else {
#pragma unroll
    for (int t = 0; t < 8; ++t) o[t] = f2bf(0.f);
  }
  *(v8bf*)(Wb + row * DIM + col) = o;
}

__global__ void k_convert_c(const float* __restrict__ src, __bf16* __restrict__ dst) {
  const size_t idx = (size_t)blockIdx.x * blockDim.x + threadIdx.x;  // 8-elem chunk
  if (idx >= (size_t)BSZ * NSENSE * SEQ * (DIM / 8)) return;
  const size_t e0 = idx * 8;
  const float4 f0 = *(const float4*)(src + e0);
  const float4 f1 = *(const float4*)(src + e0 + 4);
  v8bf o;
  o[0] = f2bf(f0.x); o[1] = f2bf(f0.y); o[2] = f2bf(f0.z); o[3] = f2bf(f0.w);
  o[4] = f2bf(f1.x); o[5] = f2bf(f1.y); o[6] = f2bf(f1.z); o[7] = f2bf(f1.w);
  *(v8bf*)(dst + e0) = o;
}

__global__ void k_gather_wg(const int* __restrict__ ids,
                            const __bf16* __restrict__ Wb,
                            __bf16* __restrict__ Wg) {
  const size_t idx = (size_t)blockIdx.x * blockDim.x + threadIdx.x;
  if (idx >= (size_t)BSZ * SEQ * (DIM / 8)) return;
  const size_t bs  = idx / (DIM / 8);
  const int    col = (int)(idx % (DIM / 8)) * 8;
  const size_t row = (size_t)ids[bs];
  *(v8bf*)(Wg + bs * DIM + col) = *(const v8bf*)(Wb + row * DIM + col);
}

__global__ void k_weights(const float* __restrict__ rowsum,
                          const int* __restrict__ ids,
                          const float* __restrict__ cwt,
                          float* __restrict__ wts) {
  const int idx = blockIdx.x * blockDim.x + threadIdx.x;
  if (idx >= BSZ * NSENSE * SEQ) return;
  const int s  = idx & (SEQ - 1);
  const int bk = idx >> 10;
  const int b  = bk >> 4;
  const int k  = bk & 15;
  const float sims = rowsum[idx];
  float sc = 1.f / (1.f + __expf(0.1f * sims - 6.f));   // sigmoid(-0.1*sims + 6)
  sc *= 1.f + (float)s * 0.01f;                          // upweight_nearby
  const float cw = cwt[(size_t)ids[b * SEQ + s] * NSENSE + k];
  wts[idx] = cw * sc + (1.f - sc);                       // [b][k][s]
}

// contentT_scaled[b,k,d,j] = content[b,k,j,d] * wts[b,k,j]
// Tile staged via async global->LDS DMA (ASYNCcnt path), weight applied at the
// transposed write (j is the column index there, so w is a per-thread scalar).
__global__ void k_transpose(const float* __restrict__ content,
                            const float* __restrict__ wts,
                            __bf16* __restrict__ cT) {
  __shared__ float tile[32][33];
  const int bk = blockIdx.y;
  const int tj = blockIdx.x % (SEQ / 32);
  const int td = blockIdx.x / (SEQ / 32);
  const int j0 = tj * 32, d0 = td * 32;
  const int tx = threadIdx.x, ty = threadIdx.y;   // (32, 8)
  const int tid = ty * 32 + tx;                   // 0..255

  // async copy: each thread moves one 16B chunk; 256 * 16B = 32x32 f32 tile
  {
    const int jj = tid >> 3;            // 0..31
    const int cc = (tid & 7) * 4;       // 0,4,...,28
    const float* gp = content + ((size_t)bk * SEQ + j0 + jj) * DIM + d0 + cc;
    unsigned long long ga = (unsigned long long)(const void*)gp;
    unsigned lo = (unsigned)(uintptr_t)(void*)&tile[jj][cc];
    asm volatile("global_load_async_to_lds_b128 %0, %1, off"
                 :: "v"(lo), "v"(ga) : "memory");
    asm volatile("s_wait_asynccnt 0x0" ::: "memory");
  }
  __syncthreads();

  const float w = wts[(size_t)bk * SEQ + j0 + tx];
#pragma unroll
  for (int r = 0; r < 4; ++r) {
    const int dd = ty + r * 8;
    cT[((size_t)bk * DIM + d0 + dd) * SEQ + j0 + tx] = f2bf(tile[tx][dd] * w);
  }
}

// ---------------- WMMA kernels ----------------

// rowsum[b,k,i] = sum_j relu( sum_d content[b,k,i,d] * Wg[b,j,d] )
__global__ void __launch_bounds__(256) k_sims(const __bf16* __restrict__ cB,
                                              const __bf16* __restrict__ Wg,
                                              float* __restrict__ rowsum) {
  const int lane  = threadIdx.x & 31;
  const int wave  = blockIdx.x * 8 + (threadIdx.x >> 5);
  const int itile = wave & 63;           // SEQ/16
  const int bk    = wave >> 6;           // b*NSENSE + k
  const int b     = bk >> 4;
  const int i0    = itile << 4;
  const int lr    = lane & 15;
  const int abase = (lane < 16) ? 0 : 8;
  const int klo   = (lane < 16) ? 0 : 8;

  const __bf16* arow = cB + ((size_t)bk * SEQ + i0 + lr) * DIM;
  const __bf16* wgb  = Wg + (size_t)b * SEQ * DIM;

  float rowAcc[8];
#pragma unroll
  for (int r = 0; r < 8; ++r) rowAcc[r] = 0.f;

  for (int g = 0; g < SEQ / 64; ++g) {   // groups of 4 j-tiles (shared A frag)
    v8f c0, c1, c2, c3;
#pragma unroll
    for (int r = 0; r < 8; ++r) { c0[r] = 0.f; c1[r] = 0.f; c2[r] = 0.f; c3[r] = 0.f; }
    const __bf16* b0 = wgb + (size_t)(g * 64 +  0 + lr) * DIM + klo;
    const __bf16* b1 = wgb + (size_t)(g * 64 + 16 + lr) * DIM + klo;
    const __bf16* b2 = wgb + (size_t)(g * 64 + 32 + lr) * DIM + klo;
    const __bf16* b3 = wgb + (size_t)(g * 64 + 48 + lr) * DIM + klo;
    for (int ks = 0; ks < DIM; ks += 32) {
      const v16bf a = load_frag_a_bf16(arow + ks + abase);
      c0 = wmma_bf16(a, load_frag_b_bf16(b0 + ks), c0);
      c1 = wmma_bf16(a, load_frag_b_bf16(b1 + ks), c1);
      c2 = wmma_bf16(a, load_frag_b_bf16(b2 + ks), c2);
      c3 = wmma_bf16(a, load_frag_b_bf16(b3 + ks), c3);
    }
#pragma unroll
    for (int r = 0; r < 8; ++r)
      rowAcc[r] += fmaxf(c0[r], 0.f) + fmaxf(c1[r], 0.f) +
                   fmaxf(c2[r], 0.f) + fmaxf(c3[r], 0.f);
  }
  // reduce across the 16 lanes of each half-wave (rows r / r+8)
#pragma unroll
  for (int m = 8; m >= 1; m >>= 1)
#pragma unroll
    for (int r = 0; r < 8; ++r)
      rowAcc[r] += __shfl_xor(rowAcc[r], m, 32);
  if ((lane & 15) == 0) {
    const int mb = (lane >> 4) << 3;
#pragma unroll
    for (int r = 0; r < 8; ++r)
      rowsum[(size_t)bk * SEQ + i0 + mb + r] = rowAcc[r];
  }
}

// hidden[b,i,d] = sum_k sum_j ctx[b,k,i,j] * cT[b,k,d,j]   (bf16 out)
// Each wave: 16 rows x 128 cols (8 N-tiles). Waves in a block share the
// d-group (B tiles hit in WGP$); consecutive i-tiles per wave.
__global__ void __launch_bounds__(256) k_hidden(const float* __restrict__ ctx,
                                                const __bf16* __restrict__ cT,
                                                __bf16* __restrict__ hid) {
  const int lane  = threadIdx.x & 31;
  const int wid   = threadIdx.x >> 5;
  const int ngrp  = blockIdx.x % (DIM / 128);      // 6
  const int rem   = blockIdx.x / (DIM / 128);
  const int ig    = rem & 7;
  const int b     = rem >> 3;
  const int itile = ig * 8 + wid;
  const int i0 = itile << 4, d0 = ngrp << 7;
  const int lr    = lane & 15;
  const int abase = (lane < 16) ? 0 : 8;
  const int klo   = (lane < 16) ? 0 : 8;

  v8f acc[8];
#pragma unroll
  for (int n = 0; n < 8; ++n)
#pragma unroll
    for (int r = 0; r < 8; ++r) acc[n][r] = 0.f;

  for (int k = 0; k < NSENSE; ++k) {
    const int bk = b * NSENSE + k;
    const float*  arow = ctx + ((size_t)bk * SEQ + i0 + lr) * SEQ;
    const __bf16* bb   = cT + ((size_t)bk * DIM + d0) * SEQ + klo;
    for (int j0 = 0; j0 < SEQ; j0 += 32) {
      const v16bf a = load_frag_a_f32(arow + j0 + abase);
#pragma unroll
      for (int n = 0; n < 8; ++n)
        acc[n] = wmma_bf16(a, load_frag_b_bf16(bb + (size_t)(n * 16 + lr) * SEQ + j0),
                           acc[n]);
    }
  }
  const size_t orow = (size_t)b * SEQ + i0 + ((lane >> 4) << 3);
#pragma unroll
  for (int n = 0; n < 8; ++n)
#pragma unroll
    for (int r = 0; r < 8; ++r)
      hid[(orow + r) * DIM + d0 + n * 16 + lr] = f2bf(acc[n][r]);
}

// logits[b,i,v] = sum_d hid[b,i,d] * W[v,d]
// Waves in a block share the v-group (W tiles hit in WGP$), differ in i-tile.
__global__ void __launch_bounds__(256) k_logits(const __bf16* __restrict__ hid,
                                                const __bf16* __restrict__ Wb,
                                                float* __restrict__ out) {
  const int lane  = threadIdx.x & 31;
  const int wid   = threadIdx.x >> 5;
  const int vgrp  = blockIdx.x % (VPAD / 64);      // 786
  const int rem   = blockIdx.x / (VPAD / 64);
  const int ig    = rem & 7;
  const int b     = rem >> 3;
  const int itile = ig * 8 + wid;
  const int i0 = itile << 4, v0 = vgrp << 6;
  const int lr    = lane & 15;
  const int abase = (lane < 16) ? 0 : 8;
  const int klo   = (lane < 16) ? 0 : 8;

  const __bf16* arow = hid + ((size_t)b * SEQ + i0 + lr) * DIM;

  v8f acc[4];
#pragma unroll
  for (int n = 0; n < 4; ++n)
#pragma unroll
    for (int r = 0; r < 8; ++r) acc[n][r] = 0.f;

  for (int ks = 0; ks < DIM; ks += 32) {
    const v16bf a = load_frag_a_bf16(arow + ks + abase);
    __builtin_prefetch((const void*)(arow + ks + 64), 0, 1);
#pragma unroll
    for (int n = 0; n < 4; ++n) {
      const __bf16* brow = Wb + (size_t)(v0 + n * 16 + lr) * DIM + ks + klo;
      acc[n] = wmma_bf16(a, load_frag_b_bf16(brow), acc[n]);
    }
  }
  const size_t irow = (size_t)b * SEQ + i0 + ((lane >> 4) << 3);
#pragma unroll
  for (int n = 0; n < 4; ++n) {
    const int v = v0 + n * 16 + lr;
    if (v < VOC) {
#pragma unroll
      for (int r = 0; r < 8; ++r)
        out[(irow + r) * (size_t)VOC + v] = acc[n][r];
    }
  }
}

// ---------------- launcher ----------------

extern "C" void kernel_launch(void* const* d_in, const int* in_sizes, int n_in,
                              void* d_out, int out_size, void* d_ws, size_t ws_size,
                              hipStream_t stream) {
  (void)in_sizes; (void)n_in; (void)out_size; (void)ws_size;
  const int*   ids     = (const int*)  d_in[0];
  const float* content = (const float*)d_in[1];
  const float* ctx     = (const float*)d_in[2];
  const float* W       = (const float*)d_in[3];
  const float* cwt     = (const float*)d_in[4];
  float* out = (float*)d_out;

  char* ws = (char*)d_ws;
  __bf16* Wb     = (__bf16*)(ws + 0ull);          //  77,266,944 B : VPAD x DIM bf16
  __bf16* Wg     = (__bf16*)(ws + 77266944ull);   //   3,145,728 B : BSZ*SEQ x DIM bf16
  __bf16* cT     = (__bf16*)(ws + 80412672ull);   //  50,331,648 B : BSZ*NSENSE*DIM x SEQ bf16
  __bf16* hid    = (__bf16*)(ws + 130744320ull);  //   3,145,728 B : BSZ*SEQ x DIM bf16
  float*  rowsum = (float*) (ws + 133890048ull);  //     131,072 B : BSZ*NSENSE*SEQ f32
  float*  wts    = (float*) (ws + 134021120ull);  //     131,072 B : BSZ*NSENSE*SEQ f32
  __bf16* cB     = (__bf16*)(ws + 134152192ull);  //  50,331,648 B : content bf16

  k_convert_w<<<(VPAD * (DIM / 8) + 255) / 256, 256, 0, stream>>>(W, Wb);
  k_convert_c<<<(BSZ * NSENSE * SEQ * (DIM / 8) + 255) / 256, 256, 0, stream>>>(content, cB);
  k_gather_wg<<<(BSZ * SEQ * (DIM / 8) + 255) / 256, 256, 0, stream>>>(ids, Wb, Wg);
  k_sims<<<BSZ * NSENSE * (SEQ / 16) / 8, 256, 0, stream>>>(cB, Wg, rowsum);
  k_weights<<<(BSZ * NSENSE * SEQ + 255) / 256, 256, 0, stream>>>(rowsum, ids, cwt, wts);
  {
    dim3 tb(32, 8, 1);
    dim3 tg((SEQ / 32) * (DIM / 32), BSZ * NSENSE, 1);
    k_transpose<<<tg, tb, 0, stream>>>(content, wts, cT);
  }
  k_hidden<<<BSZ * 8 * (DIM / 128), 256, 0, stream>>>(ctx, cT, hid);
  k_logits<<<BSZ * 8 * (VPAD / 64), 256, 0, stream>>>(hid, Wb, out);
}
